// T5Attention_11708080849455
// MI455X (gfx1250) — compile-verified, matrix operand order checked by reading, not run on hardware
//
#include <hip/hip_runtime.h>
#include <hip/hip_bf16.h>

// ---------------- types ----------------
typedef __bf16 v16bf __attribute__((ext_vector_type(16)));
typedef float  v8f   __attribute__((ext_vector_type(8)));

union FragU {            // 16 bf16 operands = 8 VGPRs, filled by two b128 loads
    uint4 u[2];
    v16bf v;
};

// ---------------- problem constants ----------------
constexpr int Bc  = 2;
constexpr int Sc  = 2048;
constexpr int Dc  = 1024;
constexpr int Hc  = 16;
constexpr int DKc = 64;
constexpr int RELSPAN = 2 * Sc - 1;   // 4095

__device__ __forceinline__ unsigned short f2bf(float f) {
    // round-to-nearest-even fp32 -> bf16
    unsigned int u = __float_as_uint(f);
    unsigned int r = u + 0x7FFFu + ((u >> 16) & 1u);
    return (unsigned short)(r >> 16);
}

__device__ __forceinline__ v8f wmma_bf16(const FragU& a, const FragU& b, v8f c) {
    return __builtin_amdgcn_wmma_f32_16x16x32_bf16(false, a.v, false, b.v,
                                                   (short)0, c, false, false);
}

// xor-butterfly lane exchange via ds_swizzle (imm pattern, no addr VGPR)
template <int IMM>
__device__ __forceinline__ float swz(float v) {
    return __int_as_float(__builtin_amdgcn_ds_swizzle(__float_as_int(v), IMM));
}
constexpr int SWAPX1 = 0x041F;
constexpr int SWAPX2 = 0x081F;
constexpr int SWAPX4 = 0x101F;
constexpr int SWAPX8 = 0x201F;

// ---------------- conversion kernels ----------------
__global__ void cvt_f32_bf16(const float* __restrict__ in,
                             unsigned short* __restrict__ out, int n) {
    int i = blockIdx.x * 256 + threadIdx.x;
    if (i < n) out[i] = f2bf(in[i]);
}

// out[n*K + k] = bf16(in[k*N + n])   (weights stored N-major for contiguous B-frags)
__global__ void transpose_cvt(const float* __restrict__ in,
                              unsigned short* __restrict__ out, int K, int N) {
    int i = blockIdx.x * 256 + threadIdx.x;
    if (i < K * N) {
        int n = i / K, k = i % K;
        out[i] = f2bf(in[k * N + n]);
    }
}

// T5 relative bucket bias table: table[h*RELSPAN + (rel + S-1)]
__global__ void bias_table_kernel(const float* __restrict__ rel_emb,
                                  float* __restrict__ table) {
    int i = blockIdx.x * 256 + threadIdx.x;
    if (i >= Hc * RELSPAN) return;
    int h   = i / RELSPAN;
    int rel = i % RELSPAN - (Sc - 1);          // rel = k - q
    int rb  = (rel > 0) ? 16 : 0;              // bidirectional, 32 buckets -> 16+16
    int ar  = rel < 0 ? -rel : rel;
    int bucket;
    if (ar < 8) {
        bucket = rb + ar;
    } else {
        // 8 + log(ar/8)/log(128/8) * 8 , clamped to 15
        float lf = __logf((float)ar * 0.125f) * (8.0f / __logf(16.0f));
        int v = 8 + (int)lf;
        bucket = rb + (v < 15 ? v : 15);
    }
    table[i] = rel_emb[bucket * Hc + h];
}

// ---------------- WMMA GEMM ----------------
// C[M,N] = A[M,K](bf16,rowmajor) x Bt[N,K](bf16,rowmajor == B^T)
// MODE 0: out bf16, Q/K layout [B,H,S,DK]
// MODE 1: out bf16, V  layout [B,H,DK,S]
// MODE 2: out fp32, rowmajor [M,N]
// block = 128 threads (4 waves); wave -> 16(M) x 64(N); grid (M/64, N/64)
template <int MODE>
__global__ void gemm_wmma(const unsigned short* __restrict__ A,
                          const unsigned short* __restrict__ Bt,
                          void* __restrict__ Cout, int M, int N, int K) {
    const int lane  = threadIdx.x & 31;
    const int wave  = threadIdx.x >> 5;
    const int mBase = blockIdx.x * 64 + wave * 16;
    const int nBase = blockIdx.y * 64;
    const int l15   = lane & 15;
    const int kHalf = (lane >> 4) * 8;        // ISA 16-bit operand k-mapping

    v8f acc[4] = {v8f{0}, v8f{0}, v8f{0}, v8f{0}};

    const uint4* Arow = (const uint4*)(A + (size_t)(mBase + l15) * K);
    for (int kc = 0; kc < K; kc += 32) {
        FragU a;
        a.u[0] = Arow[(kc + kHalf) >> 3];
        a.u[1] = Arow[(kc + kHalf + 16) >> 3];
#pragma unroll
        for (int t = 0; t < 4; ++t) {
            const uint4* Brow =
                (const uint4*)(Bt + (size_t)(nBase + t * 16 + l15) * K);
            FragU b;
            b.u[0] = Brow[(kc + kHalf) >> 3];
            b.u[1] = Brow[(kc + kHalf + 16) >> 3];
            acc[t] = wmma_bf16(a, b, acc[t]);
        }
    }

#pragma unroll
    for (int t = 0; t < 4; ++t) {
#pragma unroll
        for (int r = 0; r < 8; ++r) {
            int m = mBase + r + ((lane >> 4) << 3);   // C layout: halves hold M, M+8
            int n = nBase + t * 16 + l15;
            float v = acc[t][r];
            if (MODE == 2) {
                ((float*)Cout)[(size_t)m * N + n] = v;
            } else {
                int b = m >> 11, s = m & (Sc - 1);
                int h = n >> 6,  dk = n & (DKc - 1);
                unsigned short* o = (unsigned short*)Cout;
                if (MODE == 0)
                    o[(((size_t)(b * Hc + h)) * Sc + s) * DKc + dk] = f2bf(v);
                else
                    o[(((size_t)(b * Hc + h)) * DKc + dk) * Sc + s] = f2bf(v);
            }
        }
    }
}

// ---------------- fused flash attention ----------------
// 1 wave handles one (b,h,16-query tile); block = 4 waves; grid = B*H*(S/16)/4
__global__ void attn_kernel(const unsigned short* __restrict__ Q,
                            const unsigned short* __restrict__ Kb,
                            const unsigned short* __restrict__ Vt,   // [B,H,DK,S]
                            const float* __restrict__ bias,
                            unsigned short* __restrict__ Ctx) {      // [B,S,H*DK] bf16
    const int lane = threadIdx.x & 31;
    const int wave = threadIdx.x >> 5;
    const int gw   = blockIdx.x * 4 + wave;
    const int qt   = gw & 127;            // S/16 tiles
    const int h    = (gw >> 7) & 15;
    const int b    = gw >> 11;
    const int qBase = qt * 16;
    const int l15   = lane & 15;
    const int half8 = (lane >> 4) << 3;   // 0 or 8
    const int kHalf = half8;              // same value reused as k-offset

    __shared__ unsigned short pbuf[4][16 * 32];    // per-wave P staging (1KB)
    unsigned short* pb = pbuf[wave];

    // Q fragments for dk chunks {0..31},{32..63}, loaded once
    const uint4* Qrow =
        (const uint4*)(Q + (((size_t)(b * Hc + h)) * Sc + qBase + l15) * DKc);
    FragU qf[2];
#pragma unroll
    for (int c = 0; c < 2; ++c) {
        qf[c].u[0] = Qrow[(c * 32 + kHalf) >> 3];
        qf[c].u[1] = Qrow[(c * 32 + kHalf + 16) >> 3];
    }

    const unsigned short* Kbase = Kb + ((size_t)(b * Hc + h)) * Sc * DKc;
    const unsigned short* Vbase = Vt + ((size_t)(b * Hc + h)) * DKc * Sc;
    // per-lane bias pointer: bias element for (row r, key kb+col) = bptr[kb - r]
    const float* bptr = bias + h * RELSPAN + (Sc - 1) + l15 - qBase - half8;

    float mrow[8], lrow[8];
#pragma unroll
    for (int r = 0; r < 8; ++r) { mrow[r] = -3.0e38f; lrow[r] = 0.0f; }
    v8f ctx[4] = {v8f{0}, v8f{0}, v8f{0}, v8f{0}};

    for (int kb = 0; kb < Sc; kb += 32) {
        if (kb + 32 < Sc) {  // warm next K/V chunks into cache
            __builtin_prefetch(Kbase + (size_t)(kb + 32 + l15) * DKc, 0, 0);
            __builtin_prefetch(Vbase + (size_t)l15 * Sc + kb + 32, 0, 0);
        }

        // ---- scores: 16x32 via 2 key-subtiles x 2 dk-chunks WMMAs ----
        v8f sc[2] = {v8f{0}, v8f{0}};
#pragma unroll
        for (int kc2 = 0; kc2 < 2; ++kc2) {
            const uint4* Krow =
                (const uint4*)(Kbase + (size_t)(kb + kc2 * 16 + l15) * DKc);
#pragma unroll
            for (int c = 0; c < 2; ++c) {
                FragU kf;
                kf.u[0] = Krow[(c * 32 + kHalf) >> 3];
                kf.u[1] = Krow[(c * 32 + kHalf + 16) >> 3];
                sc[kc2] = wmma_bf16(qf[c], kf, sc[kc2]);
            }
        }

        // ---- bias add + per-row chunk max (batched butterfly stages) ----
        float mx[8];
#pragma unroll
        for (int r = 0; r < 8; ++r) {
            sc[0][r] += bptr[kb - r];
            sc[1][r] += bptr[kb + 16 - r];
            mx[r] = fmaxf(sc[0][r], sc[1][r]);
        }
        // stage-wise so the 8 ds_swizzle ops clause with one wait per stage
#pragma unroll
        for (int r = 0; r < 8; ++r) mx[r] = fmaxf(mx[r], swz<SWAPX1>(mx[r]));
#pragma unroll
        for (int r = 0; r < 8; ++r) mx[r] = fmaxf(mx[r], swz<SWAPX2>(mx[r]));
#pragma unroll
        for (int r = 0; r < 8; ++r) mx[r] = fmaxf(mx[r], swz<SWAPX4>(mx[r]));
#pragma unroll
        for (int r = 0; r < 8; ++r) mx[r] = fmaxf(mx[r], swz<SWAPX8>(mx[r]));

        // ---- online softmax update ----
        float fac[8], rs[8];
#pragma unroll
        for (int r = 0; r < 8; ++r) {
            float mnew = fmaxf(mrow[r], mx[r]);
            fac[r]   = __expf(mrow[r] - mnew);
            float p0 = __expf(sc[0][r] - mnew);
            float p1 = __expf(sc[1][r] - mnew);
            rs[r]    = p0 + p1;
            mrow[r]  = mnew;
            // stage P (C-layout -> rowmajor 16x32 bf16 in LDS)
            int mloc = r + half8;
            pb[mloc * 32 + l15]      = f2bf(p0);
            pb[mloc * 32 + 16 + l15] = f2bf(p1);
        }
#pragma unroll
        for (int r = 0; r < 8; ++r) rs[r] += swz<SWAPX1>(rs[r]);
#pragma unroll
        for (int r = 0; r < 8; ++r) rs[r] += swz<SWAPX2>(rs[r]);
#pragma unroll
        for (int r = 0; r < 8; ++r) rs[r] += swz<SWAPX4>(rs[r]);
#pragma unroll
        for (int r = 0; r < 8; ++r) rs[r] += swz<SWAPX8>(rs[r]);
#pragma unroll
        for (int r = 0; r < 8; ++r) {
            lrow[r] = lrow[r] * fac[r] + rs[r];
#pragma unroll
            for (int t = 0; t < 4; ++t) ctx[t][r] *= fac[r];
        }
        asm volatile("s_wait_dscnt 0" ::: "memory");   // LDS RAW before re-read

        // ---- reload P as A-fragment (16x32) ----
        FragU pf;
        const uint4* Prow = (const uint4*)(pb + (size_t)l15 * 32);
        pf.u[0] = Prow[kHalf >> 3];
        pf.u[1] = Prow[(kHalf + 16) >> 3];

        // ---- ctx += P @ V  (V transposed: contiguous key pairs) ----
#pragma unroll
        for (int t = 0; t < 4; ++t) {
            const uint4* Vrow =
                (const uint4*)(Vbase + (size_t)(t * 16 + l15) * Sc);
            FragU vf;
            vf.u[0] = Vrow[(kb + kHalf) >> 3];
            vf.u[1] = Vrow[(kb + kHalf + 16) >> 3];
            ctx[t] = wmma_bf16(pf, vf, ctx[t]);
        }
    }

    // ---- normalize & store ctx tile as bf16 [B,S,H*DK] ----
#pragma unroll
    for (int r = 0; r < 8; ++r) {
        float inv = 1.0f / lrow[r];
#pragma unroll
        for (int t = 0; t < 4; ++t) {
            int mloc = r + half8;
            Ctx[((size_t)(b * Sc + qBase + mloc)) * Dc + h * DKc + t * 16 + l15] =
                f2bf(ctx[t][r] * inv);
        }
    }
}

// ---------------- host-side launch ----------------
extern "C" void kernel_launch(void* const* d_in, const int* in_sizes, int n_in,
                              void* d_out, int out_size, void* d_ws, size_t ws_size,
                              hipStream_t stream) {
    const float* X    = (const float*)d_in[0];   // [B,S,D]
    const float* wq   = (const float*)d_in[1];   // [D, H*DK]
    const float* wk   = (const float*)d_in[2];
    const float* wv   = (const float*)d_in[3];
    const float* wo   = (const float*)d_in[4];   // [H*DK, D]
    const float* rel  = (const float*)d_in[5];   // [32, H]

    const size_t M = (size_t)Bc * Sc;            // 4096

    char* ws = (char*)d_ws;
    size_t off = 0;
    auto alloc = [&](size_t bytes) -> void* {
        void* p = ws + off;
        off = (off + bytes + 255) & ~(size_t)255;
        return p;
    };
    unsigned short* Xb   = (unsigned short*)alloc(M * Dc * 2);
    unsigned short* WqT  = (unsigned short*)alloc((size_t)Dc * Dc * 2);
    unsigned short* WkT  = (unsigned short*)alloc((size_t)Dc * Dc * 2);
    unsigned short* WvT  = (unsigned short*)alloc((size_t)Dc * Dc * 2);
    unsigned short* WoT  = (unsigned short*)alloc((size_t)Dc * Dc * 2);
    unsigned short* Qb   = (unsigned short*)alloc(M * Dc * 2);
    unsigned short* Kb2  = (unsigned short*)alloc(M * Dc * 2);
    unsigned short* Vb   = (unsigned short*)alloc(M * Dc * 2);
    unsigned short* Ctxb = (unsigned short*)alloc(M * Dc * 2);
    float*          bias = (float*)alloc((size_t)Hc * RELSPAN * 4);

    const int nX = (int)(M * Dc);
    cvt_f32_bf16<<<(nX + 255) / 256, 256, 0, stream>>>(X, Xb, nX);

    const int nW = Dc * Dc;
    transpose_cvt<<<(nW + 255) / 256, 256, 0, stream>>>(wq, WqT, Dc, Dc);
    transpose_cvt<<<(nW + 255) / 256, 256, 0, stream>>>(wk, WkT, Dc, Dc);
    transpose_cvt<<<(nW + 255) / 256, 256, 0, stream>>>(wv, WvT, Dc, Dc);
    transpose_cvt<<<(nW + 255) / 256, 256, 0, stream>>>(wo, WoT, Dc, Dc);

    bias_table_kernel<<<(Hc * RELSPAN + 255) / 256, 256, 0, stream>>>(rel, bias);

    dim3 gg((unsigned)(M / 64), (unsigned)(Dc / 64));
    dim3 bb(128);
    gemm_wmma<0><<<gg, bb, 0, stream>>>(Xb, WqT, Qb, (int)M, Dc, Dc);
    gemm_wmma<0><<<gg, bb, 0, stream>>>(Xb, WkT, Kb2, (int)M, Dc, Dc);
    gemm_wmma<1><<<gg, bb, 0, stream>>>(Xb, WvT, Vb, (int)M, Dc, Dc);

    // B*H*(S/16) = 4096 waves -> 1024 blocks of 4 waves
    attn_kernel<<<1024, 128, 0, stream>>>(Qb, Kb2, Vb, bias, Ctxb);

    gemm_wmma<2><<<gg, bb, 0, stream>>>(Ctxb, WoT, d_out, (int)M, Dc, Dc);
}
